// GNNStack_34961033790253
// MI455X (gfx1250) — compile-verified, matrix operand order checked by reading, not run on hardware
//
#include <hip/hip_runtime.h>

#define N_NODES 100000
#define N_EDGES 1600000
#define HDIM    128

typedef __attribute__((ext_vector_type(16))) __bf16 v16bf;
typedef __attribute__((ext_vector_type(8)))  float  v8f;

union Frag16 { v16bf v; unsigned int u[8]; };

__device__ __forceinline__ unsigned short f2bf(float f) {
    unsigned int u = __float_as_uint(f);
    unsigned int r = u + 0x7FFFu + ((u >> 16) & 1u);   // round-to-nearest-even
    return (unsigned short)(r >> 16);
}

// ---------------- degree / norm precompute ----------------

__global__ void deg_init_kernel(float* deg, int n) {
    int i = blockIdx.x * blockDim.x + threadIdx.x;
    if (i < n) deg[i] = 1.0f;                           // self loop
}

__global__ void deg_accum_kernel(const int* __restrict__ dst, float* deg, int e) {
    int i = blockIdx.x * blockDim.x + threadIdx.x;
    if (i < e) atomicAdd(&deg[dst[i]], 1.0f);
}

__global__ void deg_rsqrt_kernel(float* deg, int n) {
    int i = blockIdx.x * blockDim.x + threadIdx.x;
    if (i < n) deg[i] = rsqrtf(deg[i]);                 // deg >= 1 always
}

__global__ void edge_norm_kernel(const int* __restrict__ src, const int* __restrict__ dst,
                                 const float* __restrict__ dinv, float* nrm, int e) {
    int i = blockIdx.x * blockDim.x + threadIdx.x;
    if (i < e) nrm[i] = dinv[src[i]] * dinv[dst[i]];
}

// ---------------- WMMA GEMM: out[M x 128] = (relu?)(A[M x 128]) @ W[128 x 128] (+bias) ----------------
// 128-row M tile per block, full K=128 and N=128 in LDS (bf16), 8 waves, each wave
// computes a 16x128 strip = 8 accumulators of v_wmma_f32_16x16x32_bf16, 4 K-steps.

__global__ __launch_bounds__(256) void gemm_bf16_kernel(
        const float* __restrict__ A, const float* __restrict__ W,
        const float* __restrict__ bias, float* __restrict__ out,
        int M, int relu_in)
{
    __shared__ unsigned short lA[128 * 128];   // [m][k] row-major, 32 KB
    __shared__ unsigned short lB[128 * 128];   // [n][k] (W transposed), 32 KB

    const int tid   = threadIdx.x;
    const int mBase = blockIdx.x * 128;

    // Stage A (fp32 -> bf16, optional ReLU, zero-pad past M). 4096 float4 loads.
    for (int i = tid; i < 128 * 32; i += 256) {
        int r  = i >> 5;
        int c4 = (i & 31) << 2;
        int gr = mBase + r;
        float4 v = make_float4(0.f, 0.f, 0.f, 0.f);
        if (gr < M) v = *(const float4*)(A + (size_t)gr * HDIM + c4);
        if (relu_in) {
            v.x = fmaxf(v.x, 0.f); v.y = fmaxf(v.y, 0.f);
            v.z = fmaxf(v.z, 0.f); v.w = fmaxf(v.w, 0.f);
        }
        unsigned short* p = lA + r * 128 + c4;
        p[0] = f2bf(v.x); p[1] = f2bf(v.y); p[2] = f2bf(v.z); p[3] = f2bf(v.w);
    }
    // Stage W transposed: (k, n) -> lB[n*128 + k]
    for (int i = tid; i < 128 * 32; i += 256) {
        int k  = i >> 5;
        int n4 = (i & 31) << 2;
        float4 v = *(const float4*)(W + (size_t)k * HDIM + n4);
        lB[(n4 + 0) * 128 + k] = f2bf(v.x);
        lB[(n4 + 1) * 128 + k] = f2bf(v.y);
        lB[(n4 + 2) * 128 + k] = f2bf(v.z);
        lB[(n4 + 3) * 128 + k] = f2bf(v.w);
    }
    __syncthreads();

    const int wave = tid >> 5;
    const int lane = tid & 31;
    const int mW   = wave * 16;            // this wave's 16-row strip
    const int lm   = lane & 15;
    const int koff = (lane >> 4) * 8;      // lanes 16-31 handle K+8 (ISA A/B 16-bit layout)

    v8f acc[8];
    for (int nt = 0; nt < 8; ++nt)
        for (int j = 0; j < 8; ++j) acc[nt][j] = 0.f;

    for (int ks = 0; ks < 4; ++ks) {
        const int kb = ks * 32;
        Frag16 af;
        const unsigned int* rowA = (const unsigned int*)(lA + (mW + lm) * 128);
        #pragma unroll
        for (int p = 0; p < 8; ++p) {
            int kp = kb + koff + ((p < 4) ? 2 * p : 8 + 2 * p);   // even K pair base
            af.u[p] = rowA[kp >> 1];
        }
        #pragma unroll
        for (int nt = 0; nt < 8; ++nt) {
            Frag16 bfg;
            const unsigned int* rowB = (const unsigned int*)(lB + (nt * 16 + lm) * 128);
            #pragma unroll
            for (int p = 0; p < 8; ++p) {
                int kp = kb + koff + ((p < 4) ? 2 * p : 8 + 2 * p);
                bfg.u[p] = rowB[kp >> 1];
            }
            acc[nt] = __builtin_amdgcn_wmma_f32_16x16x32_bf16(
                          false, af.v, false, bfg.v, (short)0, acc[nt], false, false);
        }
    }

    // C/D layout: VGPR r -> row r + 8*(lane>=16), col = lane&15
    const int rowOff = (lane >> 4) * 8;
    const int col16  = lane & 15;
    for (int nt = 0; nt < 8; ++nt) {
        int col = nt * 16 + col16;
        float bv = bias ? bias[col] : 0.f;
        #pragma unroll
        for (int r = 0; r < 8; ++r) {
            int gr = mBase + mW + r + rowOff;
            if (gr < M) out[(size_t)gr * HDIM + col] = acc[nt][r] + bv;
        }
    }
}

// ---------------- aggregation ----------------

// out[i] = bias[f] + dinv[i]^2 * hlin[i]   (self-loop + bias init)
__global__ void agg_init_kernel(const float* __restrict__ hlin, const float* __restrict__ dinv,
                                const float* __restrict__ bias, float* out, int n) {
    int i = blockIdx.x * blockDim.x + threadIdx.x;
    if (i >= n * HDIM) return;
    int node = i >> 7, f = i & 127;
    float di = dinv[node];
    out[i] = bias[f] + di * di * hlin[i];
}

// one wave32 per edge; lane handles 4 features (float4 gather, 4 f32 atomics)
__global__ void agg_edges_kernel(const float* __restrict__ hlin,
                                 const int* __restrict__ src, const int* __restrict__ dst,
                                 const float* __restrict__ nrm, float* out, int e) {
    int gw = (int)((blockIdx.x * (size_t)blockDim.x + threadIdx.x) >> 5);
    int lane = threadIdx.x & 31;
    if (gw >= e) return;
    int s = src[gw], d = dst[gw];
    float w = nrm[gw];
    float4 v = *(const float4*)(hlin + (size_t)s * HDIM + lane * 4);
    float* o = out + (size_t)d * HDIM + lane * 4;
    atomicAdd(o + 0, w * v.x);
    atomicAdd(o + 1, w * v.y);
    atomicAdd(o + 2, w * v.z);
    atomicAdd(o + 3, w * v.w);
}

// ---------------- ReLU + LayerNorm (wave32 per row, H=128 => 4 elems/lane) ----------------

__global__ void relu_ln_kernel(float* __restrict__ h, const float* __restrict__ g,
                               const float* __restrict__ b, int n) {
    int gw = (int)((blockIdx.x * (size_t)blockDim.x + threadIdx.x) >> 5);
    int lane = threadIdx.x & 31;
    if (gw >= n) return;
    float* row = h + (size_t)gw * HDIM;
    float4 v = *(float4*)(row + lane * 4);
    v.x = fmaxf(v.x, 0.f); v.y = fmaxf(v.y, 0.f);
    v.z = fmaxf(v.z, 0.f); v.w = fmaxf(v.w, 0.f);
    float s = v.x + v.y + v.z + v.w;
    for (int off = 16; off > 0; off >>= 1) s += __shfl_xor(s, off, 32);
    float mu = s * (1.0f / HDIM);
    float dx = v.x - mu, dy = v.y - mu, dz = v.z - mu, dw = v.w - mu;
    float q = dx * dx + dy * dy + dz * dz + dw * dw;
    for (int off = 16; off > 0; off >>= 1) q += __shfl_xor(q, off, 32);
    float rs = rsqrtf(q * (1.0f / HDIM) + 1e-5f);
    float4 gg = *(const float4*)(g + lane * 4);
    float4 bb = *(const float4*)(b + lane * 4);
    float4 o;
    o.x = dx * rs * gg.x + bb.x; o.y = dy * rs * gg.y + bb.y;
    o.z = dz * rs * gg.z + bb.z; o.w = dw * rs * gg.w + bb.w;
    *(float4*)(row + lane * 4) = o;
}

// ---------------- head: z = t @ mp_W2 + mp_b2 ; log_softmax over C=2 (wave per node) ----------------

__global__ void head_kernel(const float* __restrict__ t, const float* __restrict__ W2,
                            const float* __restrict__ b2, float* __restrict__ out, int n) {
    int gw = (int)((blockIdx.x * (size_t)blockDim.x + threadIdx.x) >> 5);
    int lane = threadIdx.x & 31;
    if (gw >= n) return;
    const float* row = t + (size_t)gw * HDIM;
    float s0 = 0.f, s1 = 0.f;
    for (int j = lane; j < HDIM; j += 32) {
        float v = row[j];
        s0 += v * W2[j * 2 + 0];
        s1 += v * W2[j * 2 + 1];
    }
    for (int off = 16; off > 0; off >>= 1) {
        s0 += __shfl_xor(s0, off, 32);
        s1 += __shfl_xor(s1, off, 32);
    }
    if (lane == 0) {
        float z0 = s0 + b2[0], z1 = s1 + b2[1];
        float m = fmaxf(z0, z1);
        float lse = m + logf(expf(z0 - m) + expf(z1 - m));
        out[(size_t)gw * 2 + 0] = z0 - lse;
        out[(size_t)gw * 2 + 1] = z1 - lse;
    }
}

// ---------------- launch ----------------

extern "C" void kernel_launch(void* const* d_in, const int* in_sizes, int n_in,
                              void* d_out, int out_size, void* d_ws, size_t ws_size,
                              hipStream_t stream) {
    const float* x      = (const float*)d_in[0];
    const int*   ei     = (const int*)  d_in[1];   // [2, E]
    const float* W1     = (const float*)d_in[2];
    const float* b1     = (const float*)d_in[3];
    const float* W2     = (const float*)d_in[4];
    const float* b2     = (const float*)d_in[5];
    const float* W3     = (const float*)d_in[6];
    const float* b3     = (const float*)d_in[7];
    const float* ln1_g  = (const float*)d_in[8];
    const float* ln1_b  = (const float*)d_in[9];
    const float* ln2_g  = (const float*)d_in[10];
    const float* ln2_b  = (const float*)d_in[11];
    const float* mp_W1  = (const float*)d_in[12];
    const float* mp_b1  = (const float*)d_in[13];
    const float* mp_W2  = (const float*)d_in[14];
    const float* mp_b2  = (const float*)d_in[15];

    const int* src = ei;
    const int* dst = ei + N_EDGES;

    float* emb  = (float*)d_out;                    // [N, H]
    float* prob = (float*)d_out + (size_t)N_NODES * HDIM;  // [N, 2]

    float* ws   = (float*)d_ws;
    float* dinv = ws;                               // N
    float* nrm  = dinv + N_NODES;                   // E
    float* bufA = nrm + N_EDGES;                    // N*H  (lin output)
    float* bufB = bufA + (size_t)N_NODES * HDIM;    // N*H  (aggregated activations)

    const int T = 256;
    dim3 blk(T);
    dim3 gN((N_NODES + T - 1) / T);
    dim3 gE((N_EDGES + T - 1) / T);
    dim3 gNH(((size_t)N_NODES * HDIM + T - 1) / T);
    dim3 gEdgeWave((N_EDGES + 7) / 8);              // 8 waves/block, wave per edge
    dim3 gNodeWave((N_NODES + 7) / 8);              // wave per node
    dim3 gGemm((N_NODES + 127) / 128);

    // GCN normalization
    deg_init_kernel<<<gN, blk, 0, stream>>>(dinv, N_NODES);
    deg_accum_kernel<<<gE, blk, 0, stream>>>(dst, dinv, N_EDGES);
    deg_rsqrt_kernel<<<gN, blk, 0, stream>>>(dinv, N_NODES);
    edge_norm_kernel<<<gE, blk, 0, stream>>>(src, dst, dinv, nrm, N_EDGES);

    // layer 1
    gemm_bf16_kernel<<<gGemm, blk, 0, stream>>>(x, W1, nullptr, bufA, N_NODES, 0);
    agg_init_kernel<<<gNH, blk, 0, stream>>>(bufA, dinv, b1, bufB, N_NODES);
    agg_edges_kernel<<<gEdgeWave, blk, 0, stream>>>(bufA, src, dst, nrm, bufB, N_EDGES);
    relu_ln_kernel<<<gNodeWave, blk, 0, stream>>>(bufB, ln1_g, ln1_b, N_NODES);

    // layer 2
    gemm_bf16_kernel<<<gGemm, blk, 0, stream>>>(bufB, W2, nullptr, bufA, N_NODES, 0);
    agg_init_kernel<<<gNH, blk, 0, stream>>>(bufA, dinv, b2, bufB, N_NODES);
    agg_edges_kernel<<<gEdgeWave, blk, 0, stream>>>(bufA, src, dst, nrm, bufB, N_EDGES);
    relu_ln_kernel<<<gNodeWave, blk, 0, stream>>>(bufB, ln2_g, ln2_b, N_NODES);

    // layer 3 -> emb (aggregated, pre-ReLU) straight into d_out
    gemm_bf16_kernel<<<gGemm, blk, 0, stream>>>(bufB, W3, nullptr, bufA, N_NODES, 0);
    agg_init_kernel<<<gNH, blk, 0, stream>>>(bufA, dinv, b3, emb, N_NODES);
    agg_edges_kernel<<<gEdgeWave, blk, 0, stream>>>(bufA, src, dst, nrm, emb, N_EDGES);

    // head: relu(emb) @ mp_W1 + mp_b1 -> bufA ; then 128->2 + log_softmax
    gemm_bf16_kernel<<<gGemm, blk, 0, stream>>>(emb, mp_W1, mp_b1, bufA, N_NODES, 1);
    head_kernel<<<gNodeWave, blk, 0, stream>>>(bufA, mp_W2, mp_b2, prob, N_NODES);
}
